// RelevanceFilter_2061584302483
// MI455X (gfx1250) — matrix operand order repaired and checked by reading, not compiled, not run on hardware
//
#include <hip/hip_runtime.h>

// ---------------------------------------------------------------------------
// RelevanceFilter fused kernel for gfx1250 (MI455X), wave32 + WMMA bf16.
//
// Math (per batch b):
//   v[o,s]   = Wv_w[256+o,:] . x[:,s] + Wv_b[256+o]              (o in 0..255)
//   att[g,s] = w_att[b,g,:]  . x[:,s] + batt[b,g]                (g in 0..7)
//     where w_att[b,g,:] = sum_j kern[b,g*32+j] * Wv_w[g*32+j,:]
//   out[b,o,s]  = v[o,s] * sigmoid(att[o>>5, s])
//   maps[t,b,hw] = mean_g att[g,s]
//
// GEMM is 272x256 (17 M-tiles) x 256 (K) x 65536 (N) per batch, done with
// V_WMMA_F32_16X16X32_BF16 using an fp32 -> bf16 hi/lo split (3 WMMAs per
// K-step per tile: hi*hi + hi*lo + lo*hi) for near-fp32 accuracy.
// ---------------------------------------------------------------------------

typedef __attribute__((ext_vector_type(16))) __bf16 v16bf;
typedef __attribute__((ext_vector_type(8)))  float  v8f;
typedef __attribute__((ext_vector_type(8)))  int    v8i;

#define SPATIAL   65536            // T*H*W = 16*64*64
#define VD        256
#define MTILES    17               // 256 v channels + (8 att + 8 pad)
#define NBLK      128              // N columns per workgroup (8 waves * 16)

static __device__ __forceinline__ unsigned short f2bf(float f) {
    unsigned u = __float_as_uint(f);
    u += 0x7FFFu + ((u >> 16) & 1u);       // round-to-nearest-even
    return (unsigned short)(u >> 16);
}
static __device__ __forceinline__ float bf2f(unsigned short h) {
    return __uint_as_float(((unsigned)h) << 16);
}
static __device__ __forceinline__ v16bf as_bf16x16(v8i x) {
    return __builtin_bit_cast(v16bf, x);
}

// ---------------------------------------------------------------------------
// P1: kern[b,o] = text[b,:] . Wt_w[o,:] + Wt_b[o]   (4 x 256, K=512)
//     batt[b,g] = sum_j kern[b,g*32+j] * Wv_b[g*32+j]
// ---------------------------------------------------------------------------
__global__ __launch_bounds__(256) void p1_text_proj(
    const float* __restrict__ text, const float* __restrict__ Wt_w,
    const float* __restrict__ Wt_b, const float* __restrict__ Wv_b,
    float* __restrict__ kern, float* __restrict__ batt)
{
    __shared__ float ks[256];
    int b = blockIdx.x, o = threadIdx.x;
    const float* t = text + b * 512;
    const float* w = Wt_w + (size_t)o * 512;
    float s = Wt_b[o];
    for (int td = 0; td < 512; ++td) s += t[td] * w[td];
    kern[b * 256 + o] = s;
    ks[o] = s;
    __syncthreads();
    if (o < 8) {
        float bb = 0.f;
        for (int j = 0; j < 32; ++j) bb += ks[o * 32 + j] * Wv_b[o * 32 + j];
        batt[b * 8 + o] = bb;
    }
}

// ---------------------------------------------------------------------------
// P2: pre-pack A-matrix fragments (hi/lo bf16) in the ISA 16-bit A layout.
// A-fragment for (b, mt, kk): 16x32 bf16.  Lane L holds row M = L&15;
//   lane<16 : dword j -> K = (j<4 ? 2j : 16+2(j-4)), pair (K, K+1)
//   lane>=16: same + 8
// Storage: wsA[frag*512 + sel*256 + L*8 + j], frag = (b*17+mt)*8+kk, sel=hi/lo
// -> GEMM kernel reads each half-fragment as one 32-byte vector per lane.
// ---------------------------------------------------------------------------
__global__ __launch_bounds__(32) void p2_pack_afrag(
    const float* __restrict__ Wv_w, const float* __restrict__ kern,
    unsigned* __restrict__ wsA)
{
    int kk = blockIdx.x, mt = blockIdx.y, b = blockIdx.z;
    int L = threadIdx.x, m = L & 15, lh = L >> 4;
    unsigned hi[8], lo[8];
#pragma unroll
    for (int j = 0; j < 8; ++j) {
        int k  = ((j < 4) ? 2 * j : 16 + 2 * (j - 4)) + (lh ? 8 : 0);
        int c0 = kk * 32 + k;
        float a0 = 0.f, a1 = 0.f;
        if (mt < 16) {
            const float* wr = Wv_w + (size_t)(256 + mt * 16 + m) * VD;
            a0 = wr[c0]; a1 = wr[c0 + 1];
        } else if (m < 8) {
            const float* kv = kern + b * 256 + m * 32;
            for (int jj = 0; jj < 32; ++jj) {
                float kw = kv[jj];
                const float* wr = Wv_w + (size_t)(m * 32 + jj) * VD;
                a0 += kw * wr[c0]; a1 += kw * wr[c0 + 1];
            }
        }
        unsigned short h0 = f2bf(a0), h1 = f2bf(a1);
        float r0 = a0 - bf2f(h0), r1 = a1 - bf2f(h1);
        hi[j] = (unsigned)h0 | ((unsigned)f2bf(0.f + h1) & 0u) | ((unsigned)h1 << 16);
        lo[j] = (unsigned)f2bf(r0) | ((unsigned)f2bf(r1) << 16);
    }
    size_t base = ((size_t)((b * MTILES + mt) * 8 + kk)) * 512 + (size_t)L * 8;
#pragma unroll
    for (int j = 0; j < 8; ++j) {
        wsA[base + j]       = hi[j];
        wsA[base + 256 + j] = lo[j];
    }
}

// ---------------------------------------------------------------------------
// Main fused GEMM + gating kernel.
// Grid: (512 N-blocks, 4 batches), 256 threads = 8 waves.
// Wave w owns N columns [n0 + 16w, n0 + 16w + 15], all 17 M-tiles.
// ---------------------------------------------------------------------------
__global__ __launch_bounds__(256) void gemm_gate(
    const float* __restrict__ video, const float* __restrict__ Wv_b,
    const unsigned* __restrict__ wsA, const float* __restrict__ batt,
    float* __restrict__ maps, float* __restrict__ out)
{
    __shared__ float vb_s[256];
    int tid = threadIdx.x;
    vb_s[tid] = Wv_b[256 + tid];          // v-channel biases
    __syncthreads();

    int b  = blockIdx.y;
    int n0 = blockIdx.x * NBLK;
    int w  = tid >> 5, L = tid & 31;
    int lm = L & 15,  lh = L >> 4;
    int n  = n0 + w * 16 + lm;            // this lane's N column (per-batch)

    const float* xp = video + (size_t)b * VD * SPATIAL + n;

    v8f acc[MTILES] = {};

    for (int kk = 0; kk < 8; ++kk) {
        // ---- build B fragments (32x16 bf16, hi/lo) from global fp32 ----
        v8i Bhi, Blo;
#pragma unroll
        for (int j = 0; j < 8; ++j) {
            int c0 = kk * 32 + ((j < 4) ? 2 * j : 16 + 2 * (j - 4)) + (lh ? 8 : 0);
            float x0 = xp[(size_t)c0 * SPATIAL];
            float x1 = xp[(size_t)(c0 + 1) * SPATIAL];
            unsigned short h0 = f2bf(x0), h1 = f2bf(x1);
            float r0 = x0 - bf2f(h0), r1 = x1 - bf2f(h1);
            Bhi[j] = (int)((unsigned)h0 | ((unsigned)h1 << 16));
            Blo[j] = (int)((unsigned)f2bf(r0) | ((unsigned)f2bf(r1) << 16));
        }
        v16bf Bh = as_bf16x16(Bhi);
        v16bf Bl = as_bf16x16(Blo);

        const unsigned* abase =
            wsA + ((size_t)(b * MTILES) * 8 + kk) * 512 + (size_t)L * 8;
#pragma unroll
        for (int mt = 0; mt < MTILES; ++mt) {
            const unsigned* ap = abase + (size_t)mt * (8 * 512);
            v8i Ahi = *(const v8i*)(ap);
            v8i Alo = *(const v8i*)(ap + 256);
            v16bf Ah = as_bf16x16(Ahi);
            v16bf Al = as_bf16x16(Alo);
            // D += Ahi*Bhi + Ahi*Blo + Alo*Bhi  (drop lo*lo term)
            acc[mt] = __builtin_amdgcn_wmma_f32_16x16x32_bf16(
                false, Ah, false, Bh, (short)0, acc[mt], false, false);
            acc[mt] = __builtin_amdgcn_wmma_f32_16x16x32_bf16(
                false, Ah, false, Bl, (short)0, acc[mt], false, false);
            acc[mt] = __builtin_amdgcn_wmma_f32_16x16x32_bf16(
                false, Al, false, Bh, (short)0, acc[mt], false, false);
        }
    }

    // ---- epilogue: att rows live in lanes 0..15 of acc[16] (M = r = g) ----
    float sig[8];
    float msum = 0.f;
#pragma unroll
    for (int r = 0; r < 8; ++r) {
        float a = acc[16][r];
        a = __shfl(a, lm, 32);            // broadcast to upper half-wave
        a += batt[b * 8 + r];
        sig[r] = 1.f / (1.f + __expf(-a));
        msum += a;
    }

    // maps[t, b, 0, h, w] = mean_g att ; s = t*4096 + hw within this batch
    if (L < 16) {
        int t = n >> 12, hw = n & 4095;
        maps[(size_t)t * (4 * 4096) + (size_t)b * 4096 + hw] = msum * 0.125f;
    }

    // out[b, o, s] = (acc + bias) * sigmoid(att[o>>5]); gate uniform per tile
    float* op = out + (size_t)b * 256 * SPATIAL + n;
#pragma unroll
    for (int mt = 0; mt < 16; ++mt) {
        float sg = sig[mt >> 1];
#pragma unroll
        for (int r = 0; r < 8; ++r) {
            int o = mt * 16 + r + (lh ? 8 : 0);
            float val = acc[mt][r] + vb_s[o];
            op[(size_t)o * SPATIAL] = val * sg;
        }
    }
}

// ---------------------------------------------------------------------------
extern "C" void kernel_launch(void* const* d_in, const int* in_sizes, int n_in,
                              void* d_out, int out_size, void* d_ws, size_t ws_size,
                              hipStream_t stream)
{
    const float* video = (const float*)d_in[0];   // (4,256,16,64,64)
    const float* text  = (const float*)d_in[1];   // (4,1,512)
    const float* Wv_w  = (const float*)d_in[2];   // (512,256)
    const float* Wv_b  = (const float*)d_in[3];   // (512,)
    const float* Wt_w  = (const float*)d_in[4];   // (256,512)
    const float* Wt_b  = (const float*)d_in[5];   // (256,)

    float* maps = (float*)d_out;                  // (16,4,1,64,64) = 262144
    float* out  = (float*)d_out + 262144;         // (4,256,16,64,64)

    float*    kern = (float*)d_ws;                            // 1024 f32
    float*    batt = kern + 1024;                             // 32 f32
    unsigned* wsA  = (unsigned*)((char*)d_ws + 8192);         // packed A frags (~1.1 MB)

    p1_text_proj<<<4, 256, 0, stream>>>(text, Wt_w, Wt_b, Wv_b, kern, batt);
    p2_pack_afrag<<<dim3(8, MTILES, 4), 32, 0, stream>>>(Wv_w, kern, wsA);
    gemm_gate<<<dim3(SPATIAL / NBLK, 4), 256, 0, stream>>>(video, Wv_b, wsA, batt, maps, out);
}
